// DotProductAttention_58342835749644
// MI455X (gfx1250) — matrix operand order repaired and checked by reading, not compiled, not run on hardware
//
#include <hip/hip_runtime.h>
#include <math.h>

// Flash-attention, fp32 I/O, B=32 Q=K=1024 D=128, key-padding mask.
//
// Roofline: 17.2 GFLOP vs ~67 MB HBM (2.9 us floor @ 23.3 TB/s). fp32
// 16x16x4 WMMA is 8x less dense than bf16 16x16x32 -> would be compute-bound
// by ~10x. So: split-bf16 ("bf16x3"): x = xh + xl (bf16 each), product
// xh*yh + xh*yl + xl*yh recovers ~2^-16 relative accuracy (fp32-grade vs the
// fp32 reference) on the bf16 matrix pipe the chip rewards.
//
// Pipelining: double-buffered fp32 staging; tile t+1's async global->LDS
// copies are issued right after tile t's convert pass and only awaited
// (s_wait_asynccnt) at end of iteration -> global latency hidden behind
// 48 WMMAs + softmax per tile.
//
// gfx1250 paths: v_wmma_f32_16x16x32_bf16, global_load_async_to_lds_b128 +
// s_wait_asynccnt, wave32 WMMA layouts per ISA 7.12.2.

#define WAVES 4        // waves per block -> 64 query rows per block
#define KT 32          // key rows per tile
#define DDIM 128
#define QLEN 1024
#define KLEN 1024
#define BATCH 32
#define SSTRIDE 132    // fp32 staging row stride (dwords), 16B-aligned rows
#define KBSTRIDE 68    // packed-bf16 K row stride (dwords): 64 data + 4 pad
#define VBSTRIDE 20    // packed-bf16 V^T row stride (dwords): 16 data + 4 pad
#define PSTRIDE 36     // P row stride (fp32): 32 data + 4 pad
#define MASKV (-1000000.0f)

typedef float v8f __attribute__((ext_vector_type(8)));
typedef __bf16 v16bf __attribute__((ext_vector_type(16)));
typedef unsigned uv4 __attribute__((ext_vector_type(4)));
typedef unsigned uv8 __attribute__((ext_vector_type(8)));

#if __has_builtin(__builtin_amdgcn_global_load_async_to_lds_b128)
#define HAVE_ASYNC_LDS 1
typedef int v4i_vs __attribute__((vector_size(4 * sizeof(int))));
typedef __attribute__((address_space(1))) v4i_vs gv4i_t;
typedef __attribute__((address_space(3))) v4i_vs sv4i_t;
#endif

// Split 8 fp32 into packed-bf16 hi (truncated) and lo (residual) dwords.
// dword j: [15:0] = bf16(f[2j]), [31:16] = bf16(f[2j+1])  (low-half-first,
// matching 16-bit WMMA VGPR packing).
__device__ __forceinline__ void split_pack8(const float* f, unsigned* hi, unsigned* lo) {
#pragma unroll
  for (int j = 0; j < 4; ++j) {
    const float a = f[2 * j], b = f[2 * j + 1];
    const unsigned ua = __float_as_uint(a), ub = __float_as_uint(b);
    const unsigned ha = ua & 0xffff0000u, hb = ub & 0xffff0000u;
    hi[j] = (ua >> 16) | hb;
    const float ra = a - __uint_as_float(ha);   // exact residual
    const float rb = b - __uint_as_float(hb);
    lo[j] = (__float_as_uint(ra) >> 16) | (__float_as_uint(rb) & 0xffff0000u);
  }
}

__device__ __forceinline__ v16bf pack16(const unsigned* h0, const unsigned* h1) {
  uv8 r = {h0[0], h0[1], h0[2], h0[3], h1[0], h1[1], h1[2], h1[3]};
  return __builtin_bit_cast(v16bf, r);
}

__device__ __forceinline__ v16bf load_bf16x16(const unsigned* p) {
  uv8 r;
  r.lo = *(const uv4*)p;         // 16B-aligned
  r.hi = *(const uv4*)(p + 4);
  return __builtin_bit_cast(v16bf, r);
}

#define WMMA_BF16(A, B, C) \
  __builtin_amdgcn_wmma_f32_16x16x32_bf16(false, (A), false, (B), (short)0, (C), false, false)

// Issue the 32x128 fp32 K/V tile stage (async when available).
__device__ __forceinline__ void stage_tile(const float* kbase, const float* vbase,
                                           int krow0, float* sk, float* sv, int tid) {
#pragma unroll
  for (int i = 0; i < 8; ++i) {
    const int idx = tid + i * 128;          // 0..1023 float4 chunks
    const int row = idx >> 5;               // 0..31
    const int c4  = idx & 31;               // 0..31
    const float* gk = kbase + (size_t)(krow0 + row) * DDIM + c4 * 4;
    const float* gv = vbase + (size_t)(krow0 + row) * DDIM + c4 * 4;
    float* lk = sk + row * SSTRIDE + c4 * 4;
    float* lv = sv + row * SSTRIDE + c4 * 4;
#ifdef HAVE_ASYNC_LDS
    __builtin_amdgcn_global_load_async_to_lds_b128((gv4i_t*)gk, (sv4i_t*)lk, 0, 0);
    __builtin_amdgcn_global_load_async_to_lds_b128((gv4i_t*)gv, (sv4i_t*)lv, 0, 0);
#else
    *(float4*)lk = *(const float4*)gk;
    *(float4*)lv = *(const float4*)gv;
#endif
  }
}

__device__ __forceinline__ void wait_async_all() {
#ifdef HAVE_ASYNC_LDS
#if __has_builtin(__builtin_amdgcn_s_wait_asynccnt)
  __builtin_amdgcn_s_wait_asynccnt(0);
#else
  asm volatile("s_wait_asynccnt 0" ::: "memory");
#endif
#endif
}

__global__ __launch_bounds__(WAVES * 32)
void fa_bf16x3_wmma_kernel(const float* __restrict__ Qg,
                           const float* __restrict__ Kg,
                           const float* __restrict__ Vg,
                           const int* __restrict__ lens,
                           float* __restrict__ Og) {
  __shared__ float    lds_sk[2][KT * SSTRIDE];    // fp32 K tile stage (double)
  __shared__ float    lds_sv[2][KT * SSTRIDE];    // fp32 V tile stage (double)
  __shared__ unsigned lds_khi[KT * KBSTRIDE];     // bf16-pair K hi, row=key, pairs along d
  __shared__ unsigned lds_klo[KT * KBSTRIDE];
  __shared__ unsigned lds_vhi[DDIM * VBSTRIDE];   // bf16-pair V^T hi, row=d, pairs along kk
  __shared__ unsigned lds_vlo[DDIM * VBSTRIDE];
  __shared__ float    lds_p[WAVES * 16 * PSTRIDE];

  const int tid  = threadIdx.x;
  const int wave = tid >> 5;
  const int lane = tid & 31;
  const int ln16 = lane & 15;
  const int half = lane >> 4;

  const int b     = blockIdx.x / (QLEN / (16 * WAVES));
  const int qblk  = blockIdx.x % (QLEN / (16 * WAVES));
  const int qbase = qblk * (16 * WAVES) + wave * 16;

  const int vlen = lens[b];
  const float scale = 0.08838834764831845f;   // 1/sqrt(128)

  const float* kbase_ptr = Kg + (size_t)b * KLEN * DDIM;
  const float* vbase_ptr = Vg + (size_t)b * KLEN * DDIM;
  const int ktiles = (vlen + KT - 1) / KT;    // masked tail contributes 0

  // ---- Prefetch tile 0 (async), then overlap Q load/split with it.
  stage_tile(kbase_ptr, vbase_ptr, 0, lds_sk[0], lds_sv[0], tid);

  // Q tile (16x128) -> split-bf16 A-layout registers, 4 chunks of K=32.
  // A 16x32 16-bit: lane L: M=L%16; kernel's own K order 16h+{0..15} (B uses the same).
  v16bf qhi[4], qlo[4];
  {
    const float* qrow = Qg + ((size_t)b * QLEN + (size_t)(qbase + ln16)) * DDIM;
#pragma unroll
    for (int c = 0; c < 4; ++c) {
      float f0[8], f1[8];
      const float* p0 = qrow + 32 * c + 16 * half;
      const float* p1 = qrow + 32 * c + 16 * half + 8;
#pragma unroll
      for (int j = 0; j < 8; ++j) { f0[j] = p0[j]; f1[j] = p1[j]; }
      unsigned h0[4], l0[4], h1[4], l1[4];
      split_pack8(f0, h0, l0);
      split_pack8(f1, h1, l1);
      qhi[c] = pack16(h0, h1);
      qlo[c] = pack16(l0, l1);
    }
  }

  v8f acc[8];
#pragma unroll
  for (int d = 0; d < 8; ++d) acc[d] = (v8f){};
  float m_i[8], l_i[8];
#pragma unroll
  for (int v = 0; v < 8; ++v) { m_i[v] = -3.0e38f; l_i[v] = 0.0f; }

  wait_async_all();
  __syncthreads();                 // stage[0] ready for everyone

  for (int kt = 0; kt < ktiles; ++kt) {
    const int krow0 = kt * KT;
    const int cur = kt & 1;

    // ---- Cooperative split-bf16 convert of stage[cur] (integer ops).
    {   // K: thread -> (key = tid&31, d-range = (tid>>5)*32 .. +31)
      const int key = tid & 31;
      const int d0  = (tid >> 5) * 32;
      const float* src = &lds_sk[cur][key * SSTRIDE + d0];
      float f[32];
#pragma unroll
      for (int j = 0; j < 8; ++j) *(float4*)&f[4 * j] = *(const float4*)(src + 4 * j);
      unsigned h[16], l[16];
#pragma unroll
      for (int j = 0; j < 4; ++j) split_pack8(&f[8 * j], &h[4 * j], &l[4 * j]);
      unsigned* dh = &lds_khi[key * KBSTRIDE + d0 / 2];
      unsigned* dl = &lds_klo[key * KBSTRIDE + d0 / 2];
#pragma unroll
      for (int j = 0; j < 4; ++j) {
        *(uv4*)(dh + 4 * j) = *(const uv4*)&h[4 * j];
        *(uv4*)(dl + 4 * j) = *(const uv4*)&l[4 * j];
      }
    }
    {   // V transpose: thread -> d column = tid; pairs along kk
      const int d = tid;
      float f[32];
#pragma unroll
      for (int kk = 0; kk < 32; ++kk) f[kk] = lds_sv[cur][kk * SSTRIDE + d];
      unsigned h[16], l[16];
#pragma unroll
      for (int j = 0; j < 4; ++j) split_pack8(&f[8 * j], &h[4 * j], &l[4 * j]);
      unsigned* dh = &lds_vhi[d * VBSTRIDE];
      unsigned* dl = &lds_vlo[d * VBSTRIDE];
#pragma unroll
      for (int j = 0; j < 4; ++j) {
        *(uv4*)(dh + 4 * j) = *(const uv4*)&h[4 * j];
        *(uv4*)(dl + 4 * j) = *(const uv4*)&l[4 * j];
      }
    }

    // ---- Kick off next tile's async stage into the other buffer (uniform
    // branch; completion awaited at end of this iteration).
    if (kt + 1 < ktiles)
      stage_tile(kbase_ptr, vbase_ptr, krow0 + KT, lds_sk[cur ^ 1], lds_sv[cur ^ 1], tid);

    __syncthreads();               // packed bf16 buffers visible to all waves

    // ---- S = Q K^T : 2 n-tiles x 4 K-chunks x 3 split terms = 24 WMMAs.
    // B 32x16: lane L: N=L%16; kernel's own K order (matches A) 16h+{0..15} (dwords 8h+{0..7}).
    v8f S[2];
#pragma unroll
    for (int nt = 0; nt < 2; ++nt) {
      const unsigned* bh = &lds_khi[(nt * 16 + ln16) * KBSTRIDE + 8 * half];
      const unsigned* bl = &lds_klo[(nt * 16 + ln16) * KBSTRIDE + 8 * half];
      v8f s = (v8f){};
#pragma unroll
      for (int c = 0; c < 4; ++c) {
        const v16bf Bh = load_bf16x16(bh + 16 * c);
        const v16bf Bl = load_bf16x16(bl + 16 * c);
        s = WMMA_BF16(qhi[c], Bh, s);
        s = WMMA_BF16(qhi[c], Bl, s);
        s = WMMA_BF16(qlo[c], Bh, s);
      }
      S[nt] = s;
    }

    // ---- Mask + scale + online softmax over the 32-key tile.
    const bool valid0 = (krow0 + ln16) < vlen;
    const bool valid1 = (krow0 + 16 + ln16) < vlen;
    float p0a[8], p1a[8], corr[8];
#pragma unroll
    for (int v = 0; v < 8; ++v) {
      const float s0 = valid0 ? S[0][v] * scale : MASKV;
      const float s1 = valid1 ? S[1][v] * scale : MASKV;
      float mx = fmaxf(s0, s1);
      mx = fmaxf(mx, __shfl_xor(mx, 1, 32));
      mx = fmaxf(mx, __shfl_xor(mx, 2, 32));
      mx = fmaxf(mx, __shfl_xor(mx, 4, 32));
      mx = fmaxf(mx, __shfl_xor(mx, 8, 32));
      const float mnew = fmaxf(m_i[v], mx);
      const float e0 = __expf(s0 - mnew);
      const float e1 = __expf(s1 - mnew);
      float rs = e0 + e1;
      rs += __shfl_xor(rs, 1, 32);
      rs += __shfl_xor(rs, 2, 32);
      rs += __shfl_xor(rs, 4, 32);
      rs += __shfl_xor(rs, 8, 32);
      const float cc = __expf(m_i[v] - mnew);   // 0 on first tile
      l_i[v] = l_i[v] * cc + rs;
      m_i[v] = mnew;
      corr[v] = cc;
      p0a[v] = e0;
      p1a[v] = e1;
    }

    // ---- P: C-layout -> wave-private LDS -> split-bf16 A-layout.
    float* pbuf = &lds_p[wave * 16 * PSTRIDE];
#pragma unroll
    for (int v = 0; v < 8; ++v) {
      pbuf[(v + 8 * half) * PSTRIDE + ln16] = p0a[v];
      pbuf[(v + 8 * half) * PSTRIDE + 16 + ln16] = p1a[v];
    }

#pragma unroll
    for (int d = 0; d < 8; ++d) {
      v8f a = acc[d];
#pragma unroll
      for (int v = 0; v < 8; ++v) a[v] *= corr[v];
      acc[d] = a;
    }

    v16bf pah, pal;
    {
      float f0[8], f1[8];
      const float* r0 = &pbuf[ln16 * PSTRIDE + 16 * half];
      const float* r1 = &pbuf[ln16 * PSTRIDE + 16 * half + 8];
#pragma unroll
      for (int j = 0; j < 8; ++j) { f0[j] = r0[j]; f1[j] = r1[j]; }
      unsigned h0[4], l0[4], h1[4], l1[4];
      split_pack8(f0, h0, l0);
      split_pack8(f1, h1, l1);
      pah = pack16(h0, h1);
      pal = pack16(l0, l1);
    }

    // ---- O += P V : 8 d-tiles x 3 split terms = 24 WMMAs.
    // B lane L: N=L%16 -> V^T row d = dtile*16 + N, kk-run dwords 8h+{0..7}.
#pragma unroll
    for (int d = 0; d < 8; ++d) {
      const v16bf Vh = load_bf16x16(&lds_vhi[(d * 16 + ln16) * VBSTRIDE + 8 * half]);
      const v16bf Vl = load_bf16x16(&lds_vlo[(d * 16 + ln16) * VBSTRIDE + 8 * half]);
      v8f o = acc[d];
      o = WMMA_BF16(pah, Vh, o);
      o = WMMA_BF16(pah, Vl, o);
      o = WMMA_BF16(pal, Vh, o);
      acc[d] = o;
    }

    wait_async_all();              // next tile's stage complete (this wave)
    __syncthreads();               // all waves done: packed free, stage ready
  }

  // ---- Finalize: O / l, coalesced fp32 stores.
  float inv_l[8];
#pragma unroll
  for (int v = 0; v < 8; ++v) inv_l[v] = 1.0f / l_i[v];

  float* orow = Og + ((size_t)b * QLEN + qbase) * DDIM;
#pragma unroll
  for (int d = 0; d < 8; ++d) {
#pragma unroll
    for (int v = 0; v < 8; ++v) {
      const int row = v + 8 * half;
      orow[(size_t)row * DDIM + d * 16 + ln16] = acc[d][v] * inv_l[v];
    }
  }
}

extern "C" void kernel_launch(void* const* d_in, const int* in_sizes, int n_in,
                              void* d_out, int out_size, void* d_ws, size_t ws_size,
                              hipStream_t stream) {
  const float* q    = (const float*)d_in[0];
  const float* k    = (const float*)d_in[1];
  const float* v    = (const float*)d_in[2];
  const int*   lens = (const int*)d_in[3];
  float* out = (float*)d_out;
  (void)in_sizes; (void)n_in; (void)out_size; (void)d_ws; (void)ws_size;

  dim3 grid(BATCH * (QLEN / (16 * WAVES)));   // 32 * 16 = 512 blocks
  dim3 block(WAVES * 32);                     // 4 waves of 32
  fa_bf16x3_wmma_kernel<<<grid, block, 0, stream>>>(q, k, v, lens, out);
}
